// RNN_91259465105785
// MI455X (gfx1250) — matrix operand order repaired and checked by reading, not compiled, not run on hardware
//
#include <hip/hip_runtime.h>

// ---------------------------------------------------------------------------
// Problem constants (from reference)
// ---------------------------------------------------------------------------
static constexpr int SEQ  = 8192;   // sequence length S
static constexpr int FEAT = 512;    // feature size F
static constexpr int HID  = 1024;   // hidden size H
static constexpr int NL   = 4;      // layers L
static constexpr int G3   = 3 * HID;  // 3072 gate rows

// WMMA vector types (gfx1250, wave32)
typedef __attribute__((ext_vector_type(16))) __bf16        v16bf;
typedef __attribute__((ext_vector_type(8)))  float          v8f;
typedef __attribute__((ext_vector_type(16))) unsigned short v16us;
typedef __attribute__((ext_vector_type(8)))  unsigned short v8us;
typedef __attribute__((ext_vector_type(4)))  unsigned       v4u;
typedef __attribute__((ext_vector_type(8)))  unsigned       v8u;

// ---------------------------------------------------------------------------
// Workspace layout (bytes)
// ---------------------------------------------------------------------------
static constexpr size_t GI0_OFF = 0;                                        // S*3H f32
static constexpr size_t ABF_OFF = GI0_OFF + (size_t)SEQ * G3 * 4;           // S*F bf16
static constexpr size_t BBF_OFF = ABF_OFF + (size_t)SEQ * FEAT * 2;         // 3H*F bf16
static constexpr size_t HST_OFF = BBF_OFF + (size_t)G3 * FEAT * 2;          // L*2*H f32
static constexpr size_t XIN_OFF = HST_OFF + (size_t)NL * 2 * HID * 4;       // L*2*H f32
static constexpr size_t BAR_OFF = XIN_OFF + (size_t)NL * 2 * HID * 4;       // barrier counters

__device__ __forceinline__ unsigned short f32_to_bf16(float f) {
  unsigned u = __float_as_uint(f);
  u = u + 0x7FFFu + ((u >> 16) & 1u);   // round-to-nearest-even
  return (unsigned short)(u >> 16);
}

// ---------------------------------------------------------------------------
// CDNA5 async copy: cache/memory -> LDS, tracked by ASYNCcnt (no VGPR staging)
// ---------------------------------------------------------------------------
__device__ __forceinline__ void async_copy_b128(const void* gaddr, void* lds_generic) {
  // Low 32 bits of a generic pointer to __shared__ = LDS byte offset.
  unsigned lds_off = (unsigned)(unsigned long long)lds_generic;
  asm volatile("global_load_async_to_lds_b128 %0, %1, off"
               :
               : "v"(lds_off), "v"(gaddr)
               : "memory");
}

__device__ __forceinline__ void wait_async_le2() {
#if __has_builtin(__builtin_amdgcn_s_wait_asynccnt)
  __builtin_amdgcn_s_wait_asynccnt(2);
#else
  asm volatile("s_wait_asynccnt 0x2" ::: "memory");
#endif
}

__device__ __forceinline__ void wait_async_0() {
#if __has_builtin(__builtin_amdgcn_s_wait_asynccnt)
  __builtin_amdgcn_s_wait_asynccnt(0);
#else
  asm volatile("s_wait_asynccnt 0x0" ::: "memory");
#endif
}

__device__ __forceinline__ void wait_tensor_0() {
#if __has_builtin(__builtin_amdgcn_s_wait_tensorcnt)
  __builtin_amdgcn_s_wait_tensorcnt(0);
#else
  asm volatile("s_wait_tensorcnt 0x0" ::: "memory");
#endif
}

// ---------------------------------------------------------------------------
// TDM: one descriptor-driven DMA of a 2D bf16 tile (rows x cols, row stride
// `stride` elements) from global memory into LDS (row-contiguous layout).
// D# packing per CDNA5 ISA Sec 8.3/8.4. 2D -> groups 2/3
// disabled (two-operand form). Issued once per wave; EXEC is ignored.
// ---------------------------------------------------------------------------
__device__ __forceinline__ void tdm_load_2d_bf16(const void* gaddr, void* lds_generic,
                                                 unsigned cols, unsigned rows,
                                                 unsigned stride, unsigned rows_total) {
  unsigned long long ga = (unsigned long long)gaddr;
  unsigned lds_off = (unsigned)(unsigned long long)lds_generic;
  v4u g0;
  g0[0] = 1u;                                               // count=1, user mode
  g0[1] = lds_off;                                          // lds_addr (bytes)
  g0[2] = (unsigned)ga;                                     // global_addr[31:0]
  g0[3] = (unsigned)((ga >> 32) & 0x01FFFFFFu) | (2u << 30);// global_addr[56:32] | type=2
  v8u g1;
  g1[0] = 1u << 16;                 // data_size = 1 (2 bytes); no multicast/pad/iterate
  g1[1] = (FEAT & 0xFFFFu) << 16;   // tensor_dim0 = 512 (bits 79:48, low half here)
  g1[2] = (rows_total & 0xFFFFu) << 16;  // tensor_dim1[15:0] (bits 95:80)
  g1[3] = (rows_total >> 16) | (cols << 16);  // tensor_dim1[31:16] | tile_dim0
  g1[4] = rows;                     // tile_dim1 (tile_dim2 = 0)
  g1[5] = stride;                   // tensor_dim0_stride[31:0]
  g1[6] = 0u;                       // stride[47:32]=0, tensor_dim1_stride lo = 0
  g1[7] = 0u;
  asm volatile("tensor_load_to_lds %0, %1"
               :
               : "s"(g0), "s"(g1)
               : "memory");
}

// ---------------------------------------------------------------------------
// Kernel 1: fp32 -> bf16 conversion (vector x4)
// ---------------------------------------------------------------------------
__global__ void cvt_bf16_kernel(const float4* __restrict__ src,
                                ushort4* __restrict__ dst, int n4) {
  int i = blockIdx.x * 256 + threadIdx.x;
  if (i < n4) {
    float4 v = src[i];
    ushort4 o;
    o.x = f32_to_bf16(v.x); o.y = f32_to_bf16(v.y);
    o.z = f32_to_bf16(v.z); o.w = f32_to_bf16(v.w);
    dst[i] = o;
  }
}

// ---------------------------------------------------------------------------
// Kernel 2: Gi0 = batch(8192x512) * W_ih0^T(512x3072) + b_ih[0]  (bf16 WMMA)
// Block tile 128(M) x 128(N), K=512.
//  - B panel (128 x 512 bf16 = 128 KB) loaded ONCE into LDS by a single TDM
//    tensor_load_to_lds (wave 0), resident for the whole K loop (320 KB LDS).
//  - A tiles (128 x 32) double-buffered via global_load_async_to_lds_b128;
//    async loads complete in order, so ASYNCcnt<=2 guarantees the current
//    slice's 2 ops landed while the next slice's 2 stay in flight.
// 8 waves: wave_m in {0,1} (x64 rows), wave_n in {0..3} (x32 cols);
// each wave owns 4x2 = 8 WMMA 16x16 f32 accumulators.
// ---------------------------------------------------------------------------
__global__ __launch_bounds__(256) void gi0_gemm_kernel(
    const unsigned short* __restrict__ A,   // S x F  (bf16 bits, row-major, K contiguous)
    const unsigned short* __restrict__ B,   // 3H x F (bf16 bits, row-major, K contiguous)
    const float* __restrict__ bias,         // 3H = b_ih[0]
    float* __restrict__ C) {                // S x 3H
  __shared__ unsigned short sA[2][128 * 32];     //  2 x 8 KB  (A double buffer)
  __shared__ unsigned short sB[128 * FEAT];      //  128 KB    (persistent B panel)

  const int tid  = threadIdx.x;
  const int lane = tid & 31;
  const int wave = tid >> 5;
  const int bm   = blockIdx.x * 128;
  const int bn   = blockIdx.y * 128;
  const int wm   = (wave & 1) * 64;
  const int wn   = (wave >> 1) * 32;
  const int lh   = lane >> 4;   // half-wave select
  const int l16  = lane & 15;

  v8f acc[4][2];
#pragma unroll
  for (int tm = 0; tm < 4; ++tm)
#pragma unroll
    for (int tn = 0; tn < 2; ++tn)
#pragma unroll
      for (int r = 0; r < 8; ++r) acc[tm][tn][r] = 0.0f;

  // One TDM for the whole 128x512 B panel (wave 0; TDM ignores EXEC).
  if (wave == 0) {
    tdm_load_2d_bf16(&B[(size_t)bn * FEAT], &sB[0],
                     /*cols=*/FEAT, /*rows=*/128, /*stride=*/FEAT,
                     /*rows_total=*/G3);
  }

  // Stage one 128x32 K-slice of A into LDS buffer `buf` (2 async instr/wave).
  auto stageA = [&](int kt, int buf) {
#pragma unroll
    for (int i = 0; i < 2; ++i) {
      int idx = tid + i * 256;             // 0..511
      int r   = idx >> 2;                  // 0..127
      int kc  = (idx & 3) * 8;             // 0,8,16,24
      async_copy_b128(&A[(size_t)(bm + r) * FEAT + kt + kc], &sA[buf][r * 32 + kc]);
    }
  };

  stageA(0, 0);
  int buf = 0;
  for (int kt = 0; kt < FEAT; kt += 32) {
    if (kt + 32 < FEAT) {
      stageA(kt + 32, buf ^ 1);  // prefetch next slice into other buffer
      wait_async_le2();          // current slice (2 older async ops) has landed
    } else {
      wait_async_0();
    }
    if (kt == 0) wait_tensor_0();  // B panel DMA complete (nop for waves 1..7)
    __syncthreads();

    // A fragment (16x32 bf16): lanes 0-15 -> M=lane, K in {0..7,16..23};
    //                          lanes 16-31 -> M=lane-16, K in {8..15,24..31}.
    v16bf afrag[4];
#pragma unroll
    for (int tm = 0; tm < 4; ++tm) {
      const unsigned short* p = &sA[buf][(wm + tm * 16 + l16) * 32 + lh * 8];
      v8us lo = *(const v8us*)p;
      v8us hi = *(const v8us*)(p + 16);
      v16us t;
#pragma unroll
      for (int e = 0; e < 8; ++e) { t[e] = lo[e]; t[e + 8] = hi[e]; }
      afrag[tm] = __builtin_bit_cast(v16bf, t);
    }
    // B fragment (32x16 bf16) from the persistent panel (row stride 512):
    // lanes 0-15 -> N=lane, K=kt+{0..7,16..23}; lanes 16-31 -> N=lane-16, K=kt+{8..15,24..31}.
    v16bf bfrag[2];
#pragma unroll
    for (int tn = 0; tn < 2; ++tn) {
      const unsigned short* p = &sB[(wn + tn * 16 + l16) * FEAT + kt + lh * 8];
      v8us lo = *(const v8us*)p;
      v8us hi = *(const v8us*)(p + 16);
      v16us t;
#pragma unroll
      for (int e = 0; e < 8; ++e) { t[e] = lo[e]; t[e + 8] = hi[e]; }
      bfrag[tn] = __builtin_bit_cast(v16bf, t);
    }

#pragma unroll
    for (int tm = 0; tm < 4; ++tm)
#pragma unroll
      for (int tn = 0; tn < 2; ++tn)
        acc[tm][tn] = __builtin_amdgcn_wmma_f32_16x16x32_bf16(
            false, afrag[tm], false, bfrag[tn], (short)0, acc[tm][tn], false, false);

    __syncthreads();   // all waves done reading sA[buf] before it is restaged
    buf ^= 1;
  }

  // Epilogue: C/D layout (VGPR r: lanes 0-15 M=r, lanes 16-31 M=8+r; N=lane%16)
#pragma unroll
  for (int tm = 0; tm < 4; ++tm) {
#pragma unroll
    for (int tn = 0; tn < 2; ++tn) {
      int n = bn + wn + tn * 16 + l16;
      float bv = bias[n];
#pragma unroll
      for (int r = 0; r < 8; ++r) {
        int m = bm + wm + tm * 16 + r + lh * 8;
        C[(size_t)m * G3 + n] = acc[tm][tn][r] + bv;
      }
    }
  }
}

// ---------------------------------------------------------------------------
// Kernel 3: init double-buffered states + barrier counters
// ---------------------------------------------------------------------------
__global__ void seq_init_kernel(const float* __restrict__ h0,
                                float* __restrict__ hstate,
                                float* __restrict__ xin,
                                unsigned* __restrict__ bar) {
  int i = blockIdx.x * 256 + threadIdx.x;
  if (i < NL * HID) {
    int l = i / HID, u = i % HID;
    float v = h0[i];
    hstate[(size_t)(l * 2 + 0) * HID + u] = v;
    hstate[(size_t)(l * 2 + 1) * HID + u] = v;
    xin[(size_t)(l * 2 + 0) * HID + u] = 0.0f;
    xin[(size_t)(l * 2 + 1) * HID + u] = 0.0f;
  }
  if (blockIdx.x == 0 && threadIdx.x < 8) bar[threadIdx.x] = 0u;
}

// ---------------------------------------------------------------------------
// Grid-wide sense barrier (all NB blocks must be co-resident; NB=65)
// ---------------------------------------------------------------------------
static constexpr int LBLK = 16;               // blocks per layer (1024/64 units)
static constexpr int NB   = NL * LBLK + 1;    // 65 (last block = MLP head)

__device__ __forceinline__ void grid_sync(unsigned* cnt, unsigned* gen, unsigned target) {
  __syncthreads();
  if (threadIdx.x == 0) {
    __threadfence();
    unsigned prev = atomicAdd(cnt, 1u);
    if (prev == (unsigned)(NB - 1)) {
      *cnt = 0u;
      __threadfence();
      atomicAdd(gen, 1u);
    } else {
      while (*(volatile unsigned*)gen < target) { __builtin_amdgcn_s_sleep(1); }
    }
    __threadfence();
  }
  __syncthreads();
}

// ---------------------------------------------------------------------------
// Kernel 4: persistent pipelined GRU over 8192 steps.
// Global step g: layer l processes t = g - l; head processes t = g - 4.
// hstate[l][p], xin[l][p] double-buffered on step parity.
// Block = 256 threads: 64 hidden units x 4 lanes (each lane: 256-wide K slice).
// Weights (~88 MB fp32) are L2-resident (192 MB L2); HBM would need 720 TB.
// ---------------------------------------------------------------------------
__global__ __launch_bounds__(256) void gru_seq_kernel(
    const float* __restrict__ Gi0,        // S x 3H (bias already added)
    const float* __restrict__ Wih_rest,   // (L-1) x 3H x H
    const float* __restrict__ Whh,        // L x 3H x H
    const float* __restrict__ b_ih,       // L x 3H
    const float* __restrict__ b_hh,       // L x 3H
    const float* __restrict__ fc1_w,      // 32 x H
    const float* __restrict__ fc1_b,      // 32
    const float* __restrict__ fc2_w,      // 32
    const float* __restrict__ fc2_b,      // 1
    float* __restrict__ hstate,           // L x 2 x H
    float* __restrict__ xin,              // L x 2 x H
    unsigned* __restrict__ bar,           // [0]=count [1]=generation
    float* __restrict__ out) {            // S
  const int tid  = threadIdx.x;
  const int blk  = blockIdx.x;
  const bool is_head = (blk == NB - 1);
  const int layer = is_head ? 0 : (blk / LBLK);
  const int slice = is_head ? 0 : (blk % LBLK);

  __shared__ float sh[HID];
  __shared__ float sx[HID];
  __shared__ float shead[32];

  const float* Wih_l = (layer > 0) ? (Wih_rest + (size_t)(layer - 1) * G3 * HID) : nullptr;
  const float* Whh_l = Whh + (size_t)layer * G3 * HID;

  const int unit = tid >> 2;           // 0..63
  const int q    = tid & 3;            // 0..3
  const int u    = slice * 64 + unit;  // global hidden unit
  const int kq   = q * 256;            // K slice base

  const int GSTEPS = SEQ + NL;         // g = 0 .. S+3
  for (int g = 0; g < GSTEPS; ++g) {
    const int rb = g & 1;
    const int wb = rb ^ 1;
    if (!is_head) {
      const int t = g - layer;
      if (t >= 0 && t < SEQ) {
        // Stage h_prev (and x for layers > 0) into LDS.
        const float* hprev = hstate + (size_t)(layer * 2 + rb) * HID;
        for (int i = tid; i < HID; i += 256) sh[i] = hprev[i];
        if (layer > 0) {
          const float* xp = xin + (size_t)(layer * 2 + wb) * HID;  // written by l-1 at step g-1
          for (int i = tid; i < HID; i += 256) sx[i] = xp[i];
        }
        __syncthreads();

        float hr = 0.f, hz = 0.f, hn = 0.f;   // Whh dots
        float ar = 0.f, az = 0.f, an = 0.f;   // Wih dots (layers > 0)
        {
          const float* wr = Whh_l + (size_t)u * HID + kq;
          const float* wz = Whh_l + (size_t)(u + HID) * HID + kq;
          const float* wn = Whh_l + (size_t)(u + 2 * HID) * HID + kq;
#pragma unroll 8
          for (int k = 0; k < 256; k += 4) {
            float4 hv = *(const float4*)&sh[kq + k];
            float4 a = *(const float4*)&wr[k];
            float4 b = *(const float4*)&wz[k];
            float4 c = *(const float4*)&wn[k];
            hr += a.x * hv.x + a.y * hv.y + a.z * hv.z + a.w * hv.w;
            hz += b.x * hv.x + b.y * hv.y + b.z * hv.z + b.w * hv.w;
            hn += c.x * hv.x + c.y * hv.y + c.z * hv.z + c.w * hv.w;
          }
        }
        if (layer > 0) {
          const float* vr = Wih_l + (size_t)u * HID + kq;
          const float* vz = Wih_l + (size_t)(u + HID) * HID + kq;
          const float* vn = Wih_l + (size_t)(u + 2 * HID) * HID + kq;
#pragma unroll 8
          for (int k = 0; k < 256; k += 4) {
            float4 xv = *(const float4*)&sx[kq + k];
            float4 a = *(const float4*)&vr[k];
            float4 b = *(const float4*)&vz[k];
            float4 c = *(const float4*)&vn[k];
            ar += a.x * xv.x + a.y * xv.y + a.z * xv.z + a.w * xv.w;
            az += b.x * xv.x + b.y * xv.y + b.z * xv.z + b.w * xv.w;
            an += c.x * xv.x + c.y * xv.y + c.z * xv.z + c.w * xv.w;
          }
        }
        // Reduce across the 4 lanes of this unit (wave32 shuffles).
        hr += __shfl_xor(hr, 1); hr += __shfl_xor(hr, 2);
        hz += __shfl_xor(hz, 1); hz += __shfl_xor(hz, 2);
        hn += __shfl_xor(hn, 1); hn += __shfl_xor(hn, 2);
        ar += __shfl_xor(ar, 1); ar += __shfl_xor(ar, 2);
        az += __shfl_xor(az, 1); az += __shfl_xor(az, 2);
        an += __shfl_xor(an, 1); an += __shfl_xor(an, 2);

        if (q == 0) {
          float gi_r, gi_z, gi_n;
          if (layer == 0) {
            const float* gi = Gi0 + (size_t)t * G3;
            gi_r = gi[u]; gi_z = gi[u + HID]; gi_n = gi[u + 2 * HID];
          } else {
            gi_r = ar + b_ih[(size_t)layer * G3 + u];
            gi_z = az + b_ih[(size_t)layer * G3 + HID + u];
            gi_n = an + b_ih[(size_t)layer * G3 + 2 * HID + u];
          }
          float gh_r = hr + b_hh[(size_t)layer * G3 + u];
          float gh_z = hz + b_hh[(size_t)layer * G3 + HID + u];
          float gh_n = hn + b_hh[(size_t)layer * G3 + 2 * HID + u];
          float r = 1.f / (1.f + __expf(-(gi_r + gh_r)));
          float z = 1.f / (1.f + __expf(-(gi_z + gh_z)));
          float n = tanhf(gi_n + r * gh_n);
          float hnew = (1.f - z) * n + z * sh[u];
          hstate[(size_t)(layer * 2 + wb) * HID + u] = hnew;
          if (layer < NL - 1) xin[(size_t)((layer + 1) * 2 + rb) * HID + u] = hnew;
        }
      }
    } else {
      // Head: y[t] = fc2 @ relu(fc1 @ h_top(t) + b1) + b2, t = g - 4.
      const int t = g - NL;
      if (t >= 0) {
        const float* htop = hstate + (size_t)(3 * 2 + rb) * HID;  // h_top(t) lives in parity g&1
        const int o = tid >> 3, q8 = tid & 7;                     // 32 outputs x 8 lanes
        const float* w = fc1_w + (size_t)o * HID + q8 * 128;
        const float* hv = htop + q8 * 128;
        float acc = 0.f;
#pragma unroll 8
        for (int k = 0; k < 128; k += 4) {
          float4 a = *(const float4*)&w[k];
          float4 b = *(const float4*)&hv[k];
          acc += a.x * b.x + a.y * b.y + a.z * b.z + a.w * b.w;
        }
        acc += __shfl_xor(acc, 1); acc += __shfl_xor(acc, 2); acc += __shfl_xor(acc, 4);
        if (q8 == 0) shead[o] = fmaxf(acc + fc1_b[o], 0.f);
        __syncthreads();
        if (tid == 0) {
          float y = fc2_b[0];
#pragma unroll
          for (int j = 0; j < 32; ++j) y += fc2_w[j] * shead[j];
          out[t] = y;
        }
      }
    }
    grid_sync(&bar[0], &bar[1], (unsigned)(g + 1));
  }
}

// ---------------------------------------------------------------------------
// Host launcher
// ---------------------------------------------------------------------------
extern "C" void kernel_launch(void* const* d_in, const int* in_sizes, int n_in,
                              void* d_out, int out_size, void* d_ws, size_t ws_size,
                              hipStream_t stream) {
  (void)in_sizes; (void)n_in; (void)out_size; (void)ws_size;
  const float* batch     = (const float*)d_in[0];
  const float* h0        = (const float*)d_in[1];
  const float* W_ih0     = (const float*)d_in[2];
  const float* W_ih_rest = (const float*)d_in[3];
  const float* W_hh      = (const float*)d_in[4];
  const float* b_ih      = (const float*)d_in[5];
  const float* b_hh      = (const float*)d_in[6];
  const float* fc1_w     = (const float*)d_in[7];
  const float* fc1_b     = (const float*)d_in[8];
  const float* fc2_w     = (const float*)d_in[9];
  const float* fc2_b     = (const float*)d_in[10];

  char* ws = (char*)d_ws;
  float*          Gi0  = (float*)(ws + GI0_OFF);
  unsigned short* Abf  = (unsigned short*)(ws + ABF_OFF);
  unsigned short* Bbf  = (unsigned short*)(ws + BBF_OFF);
  float*          hst  = (float*)(ws + HST_OFF);
  float*          xinb = (float*)(ws + XIN_OFF);
  unsigned*       bar  = (unsigned*)(ws + BAR_OFF);

  // 1) fp32 -> bf16 conversions
  {
    int n4 = SEQ * FEAT / 4;
    cvt_bf16_kernel<<<(n4 + 255) / 256, 256, 0, stream>>>(
        (const float4*)batch, (ushort4*)Abf, n4);
    n4 = G3 * FEAT / 4;
    cvt_bf16_kernel<<<(n4 + 255) / 256, 256, 0, stream>>>(
        (const float4*)W_ih0, (ushort4*)Bbf, n4);
  }

  // 2) Layer-0 input projections: WMMA GEMM (8192x3072, K=512), bias fused
  {
    dim3 grid(SEQ / 128, G3 / 128);   // 64 x 24
    gi0_gemm_kernel<<<grid, 256, 0, stream>>>(Abf, Bbf, b_ih, Gi0);
  }

  // 3) Init double-buffered states + barrier
  seq_init_kernel<<<(NL * HID + 255) / 256, 256, 0, stream>>>(h0, hst, xinb, bar);

  // 4) Persistent pipelined GRU + head (one grid barrier per global step)
  gru_seq_kernel<<<NB, 256, 0, stream>>>(
      Gi0, W_ih_rest, W_hh, b_ih, b_hh, fc1_w, fc1_b, fc2_w, fc2_b,
      hst, xinb, bar, (float*)d_out);
}
